// Attention_3762391351877
// MI455X (gfx1250) — compile-verified, matrix operand order checked by reading, not run on hardware
//
#include <hip/hip_runtime.h>
#include <hip/hip_bf16.h>

typedef __attribute__((ext_vector_type(16))) _Float16 v16h;
typedef __attribute__((ext_vector_type(8)))  _Float16 v8h;
typedef __attribute__((ext_vector_type(8)))  float    v8f;
typedef __attribute__((ext_vector_type(4)))  float    v4f;
typedef int v4i __attribute__((__vector_size__(16)));

#define WMMA_F16(a, b, c) __builtin_amdgcn_wmma_f32_16x16x32_f16( \
    false, (a), false, (b), (short)0, (c), false, false)

// ---- gfx1250 async global->LDS (ASYNCcnt path), guarded --------------------
#if defined(__has_builtin)
#if __has_builtin(__builtin_amdgcn_global_load_async_to_lds_b128)
#define HAVE_ASYNC_LDS 1
#endif
#endif
#ifndef HAVE_ASYNC_LDS
#define HAVE_ASYNC_LDS 0
#endif

#if defined(__has_builtin) && __has_builtin(__builtin_amdgcn_s_wait_asynccnt)
#define WAIT_ASYNC(n) __builtin_amdgcn_s_wait_asynccnt(n)
#else
#define WAIT_ASYNC(n) asm volatile("s_wait_asynccnt %0" ::"i"(n))
#endif

// builtin wants: (v4i AS(1)*, v4i AS(3)*, imm offset, imm cpol)
#define GPTR(p) ((__attribute__((address_space(1))) v4i*)(p))
#define LPTR(p) ((__attribute__((address_space(3))) v4i*)(p))

constexpr int Bsz = 8, Ssz = 16, Dsz = 4096, Hn = 32, HDsz = 128, Tsz = 2048;
constexpr int Mrows = Bsz * Ssz;                 // 128
constexpr size_t OUT_ELEMS   = (size_t)Mrows * Dsz;            // 524288
constexpr size_t CACHE_ELEMS = (size_t)Bsz * Tsz * Hn * HDsz;  // 67108864

// ---- WMMA fragment helpers (wave32; ISA 7.12.2 layouts) --------------------
// A (and symmetric B): lane m=lane&15; K runs {klo..klo+7} and {16+klo..23+klo},
// klo = 8 for upper half-wave.
__device__ __forceinline__ v16h frag_from_f32(const float* rb, int k0) {
  const int klo = (threadIdx.x & 16) ? 8 : 0;
  v4f a0 = *reinterpret_cast<const v4f*>(rb + k0 + klo);
  v4f a1 = *reinterpret_cast<const v4f*>(rb + k0 + klo + 4);
  v4f a2 = *reinterpret_cast<const v4f*>(rb + k0 + klo + 16);
  v4f a3 = *reinterpret_cast<const v4f*>(rb + k0 + klo + 20);
  v16h f;
#pragma unroll
  for (int j = 0; j < 4; ++j) {
    f[j]      = (_Float16)a0[j];
    f[4 + j]  = (_Float16)a1[j];
    f[8 + j]  = (_Float16)a2[j];
    f[12 + j] = (_Float16)a3[j];
  }
  return f;
}

__device__ __forceinline__ v16h frag_from_f16(const _Float16* rb, int k0) {
  const int klo = (threadIdx.x & 16) ? 8 : 0;
  v8h lo = *reinterpret_cast<const v8h*>(rb + k0 + klo);
  v8h hi = *reinterpret_cast<const v8h*>(rb + k0 + klo + 16);
  v16h f;
#pragma unroll
  for (int j = 0; j < 8; ++j) { f[j] = lo[j]; f[8 + j] = hi[j]; }
  return f;
}

// Column gather (for V as WMMA-B: B[k][n] = V[t=k][d=n], stride = H*HD floats)
__device__ __forceinline__ v16h frag_gather_stride(const float* __restrict__ base,
                                                   int k0, int stride) {
  const int klo = (threadIdx.x & 16) ? 8 : 0;
  v16h f;
#pragma unroll
  for (int j = 0; j < 8; ++j)
    f[j] = (_Float16)base[(size_t)(k0 + klo + j) * stride];
#pragma unroll
  for (int j = 0; j < 8; ++j)
    f[8 + j] = (_Float16)base[(size_t)(k0 + 16 + klo + j) * stride];
  return f;
}

// ---- GEMM: C[M=128,N] = A[128,K] @ W[N,K]^T  (fp32 in/out, f16 WMMA) -------
// Block = 256 threads = 8 waves; wave w owns M-tile w; block owns one 16-wide
// N tile. Weight tile (16x64 f32 = 4KB) is staged straight into LDS with
// GLOBAL_LOAD_ASYNC_TO_LDS_B128 (one b128 per lane), double-buffered so the
// next tile is in flight while this one feeds 2 WMMAs.
__global__ void gemm_xwT_wmma(const float* __restrict__ A, const float* __restrict__ W,
                              float* __restrict__ C, int N, int K) {
  __shared__ __align__(16) float Bs[2][16 * 64];
  const int tid  = threadIdx.x;
  const int wid  = tid >> 5;
  const int lane = tid & 31;
  const int n0   = blockIdx.x * 16;
  const int m    = (wid << 4) + (lane & 15);
  const float* arow = A + (size_t)m * K;
  // staging: thread t -> W row n0 + (t>>4), cols (t&15)*4 .. +3 of the K tile
  const float* wsrc = W + (size_t)(n0 + (tid >> 4)) * K + ((tid & 15) << 2);
  v8f acc = {};
#if HAVE_ASYNC_LDS
  __builtin_amdgcn_global_load_async_to_lds_b128(GPTR(wsrc), LPTR(&Bs[0][tid << 2]), 0, 0);
#endif
  for (int k0 = 0; k0 < K; k0 += 64) {
    const int cur = (k0 >> 6) & 1;
    __syncthreads();   // prior consumers done before the other buffer is overwritten
#if HAVE_ASYNC_LDS
    if (k0 + 64 < K) {
      __builtin_amdgcn_global_load_async_to_lds_b128(
          GPTR(wsrc + k0 + 64), LPTR(&Bs[cur ^ 1][tid << 2]), 0, 0);
      WAIT_ASYNC(1);   // current tile landed; next still in flight
    } else {
      WAIT_ASYNC(0);
    }
#else
    *reinterpret_cast<v4f*>(&Bs[cur][tid << 2]) =
        *reinterpret_cast<const v4f*>(wsrc + k0);
    if (k0 + 64 < K) __builtin_prefetch(wsrc + k0 + 64, 0, 1);
#endif
    __syncthreads();
    const float* brow = &Bs[cur][(lane & 15) << 6];
    v16h a0 = frag_from_f32(arow, k0);
    v16h b0 = frag_from_f32(brow, 0);
    acc = WMMA_F16(a0, b0, acc);
    v16h a1 = frag_from_f32(arow, k0 + 32);
    v16h b1 = frag_from_f32(brow, 32);
    acc = WMMA_F16(a1, b1, acc);
  }
  const int row = (wid << 4) + ((lane >> 4) << 3);
  const int col = n0 + (lane & 15);
#pragma unroll
  for (int r = 0; r < 8; ++r) C[(size_t)(row + r) * N + col] = acc[r];
}

// ---- RoPE (interleaved-pair form) on xq and xk -----------------------------
__global__ void rope_kernel(float* __restrict__ xq, float* __restrict__ xk,
                            const float* __restrict__ cosb, const float* __restrict__ sinb) {
  const int i = blockIdx.x * 256 + threadIdx.x;       // 2 * 262144 threads
  constexpr int PAIRS = Mrows * Dsz / 2;              // 262144
  if (i >= 2 * PAIRS) return;
  float* buf = (i >= PAIRS) ? xk : xq;
  const int p   = (i >= PAIRS) ? i - PAIRS : i;
  const int row = p >> 11;            // 2048 pairs per row of 4096
  const int rem = p & 2047;
  const int h   = rem >> 6;
  const int ii  = rem & 63;
  const int s   = row & 15;
  const float c = cosb[s * 64 + ii], sn = sinb[s * 64 + ii];
  float* ptr = buf + (size_t)row * Dsz + h * HDsz + 2 * ii;
  const float t0 = ptr[0], t1 = ptr[1];
  ptr[0] = t0 * c - t1 * sn;
  ptr[1] = t0 * sn + t1 * c;
}

// ---- Cache passthrough: pure streaming, non-temporal so the 1 GB stream ----
// does not evict the L2-resident weights/activations that the WMMA kernels reuse.
__global__ void copy_caches(const v4f* __restrict__ sk, const v4f* __restrict__ sv,
                            v4f* __restrict__ dk, v4f* __restrict__ dv, long nf4) {
  const long stride = (long)gridDim.x * blockDim.x;
  for (long i = (long)blockIdx.x * blockDim.x + threadIdx.x; i < nf4; i += stride) {
    __builtin_nontemporal_store(__builtin_nontemporal_load(sk + i), dk + i);
    __builtin_nontemporal_store(__builtin_nontemporal_load(sv + i), dv + i);
  }
}

__global__ void scatter_kv(const v4f* __restrict__ xk, const v4f* __restrict__ xv,
                           const int* __restrict__ idx,
                           v4f* __restrict__ kout, v4f* __restrict__ vout) {
  const int i = blockIdx.x * 256 + threadIdx.x;       // B*S*H*HD/4 = 131072
  if (i >= Bsz * Ssz * Hn * HDsz / 4) return;
  const int d4 = i & 31;
  const int h  = (i >> 5) & 31;
  const int s  = (i >> 10) & 15;
  const int b  = i >> 14;
  const int t  = idx[s];
  const size_t src = (size_t)(b * Ssz + s) * (Dsz / 4) + h * (HDsz / 4) + d4;
  const size_t dst = (size_t)(b * Tsz + t) * (Hn * HDsz / 4) + h * (HDsz / 4) + d4;
  kout[dst] = xk[src];
  vout[dst] = xv[src];
}

// ---- scores[bh,s,t] = (q . k)/sqrt(HD) + mask[s,t]  via WMMA ---------------
// grid (T/64, B*H), block 128 (4 waves, one 16-wide t-tile per wave)
__global__ void attn_scores_wmma(const float* __restrict__ xq, const float* __restrict__ kcache,
                                 const float* __restrict__ mask, float* __restrict__ scores) {
  const int bh = blockIdx.y;
  const int b = bh >> 5, h = bh & 31;
  const int lane = threadIdx.x & 31, w = threadIdx.x >> 5;
  const int t0 = (blockIdx.x * 4 + w) * 16;
  const float* aq = xq + (size_t)(b * Ssz + (lane & 15)) * Dsz + h * HDsz;
  const float* bk = kcache + ((size_t)(b * Tsz + t0 + (lane & 15)) * Hn + h) * HDsz;
  v8f acc = {};
#pragma unroll
  for (int k0 = 0; k0 < HDsz; k0 += 32) {
    v16h a  = frag_from_f32(aq, k0);
    v16h bb = frag_from_f32(bk, k0);
    acc = WMMA_F16(a, bb, acc);
  }
  const float sc = 0.08838834764831845f;  // 1/sqrt(128)
  const int rbase = (lane >> 4) << 3;
  const int tc = t0 + (lane & 15);
#pragma unroll
  for (int r = 0; r < 8; ++r) {
    const int row = rbase + r;
    scores[((size_t)bh * Ssz + row) * Tsz + tc] = acc[r] * sc + mask[row * Tsz + tc];
  }
}

// ---- row softmax over T=2048, fp32 in -> f16 probs out ---------------------
__global__ void softmax_rows(const float* __restrict__ scores, _Float16* __restrict__ probs) {
  __shared__ float red[256];
  const int tid = threadIdx.x;
  const float* src = scores + (size_t)blockIdx.x * Tsz;
  float vals[8];
  float mx = -INFINITY;
#pragma unroll
  for (int j = 0; j < 8; ++j) { vals[j] = src[tid + j * 256]; mx = fmaxf(mx, vals[j]); }
  red[tid] = mx; __syncthreads();
  for (int s = 128; s > 0; s >>= 1) {
    if (tid < s) red[tid] = fmaxf(red[tid], red[tid + s]);
    __syncthreads();
  }
  mx = red[0]; __syncthreads();
  float sum = 0.f;
#pragma unroll
  for (int j = 0; j < 8; ++j) { vals[j] = __expf(vals[j] - mx); sum += vals[j]; }
  red[tid] = sum; __syncthreads();
  for (int s = 128; s > 0; s >>= 1) {
    if (tid < s) red[tid] += red[tid + s];
    __syncthreads();
  }
  const float inv = 1.0f / red[0];
  _Float16* dst = probs + (size_t)blockIdx.x * Tsz;
#pragma unroll
  for (int j = 0; j < 8; ++j) dst[tid + j * 256] = (_Float16)(vals[j] * inv);
}

// ---- out[b,s,h*HD+d] = probs @ V  via WMMA ---------------------------------
// grid B*H, block 128 (4 waves); each wave owns d-tiles {w, w+4}
__global__ void attn_pv_wmma(const _Float16* __restrict__ probs, const float* __restrict__ vcache,
                             float* __restrict__ attn) {
  const int bh = blockIdx.x;
  const int b = bh >> 5, h = bh & 31;
  const int lane = threadIdx.x & 31, w = threadIdx.x >> 5;
  const int n = lane & 15;
  const _Float16* ap = probs + ((size_t)bh * Ssz + n) * Tsz;
  const float* vb = vcache + (size_t)b * Tsz * Hn * HDsz + h * HDsz;
  const int d0 = w * 16, d1 = (w + 4) * 16;
  v8f acc0 = {}, acc1 = {};
  for (int k0 = 0; k0 < Tsz; k0 += 32) {
    v16h a  = frag_from_f16(ap, k0);
    v16h b0 = frag_gather_stride(vb + d0 + n, k0, Hn * HDsz);
    acc0 = WMMA_F16(a, b0, acc0);
    v16h b1 = frag_gather_stride(vb + d1 + n, k0, Hn * HDsz);
    acc1 = WMMA_F16(a, b1, acc1);
  }
  const int rbase = (lane >> 4) << 3;
#pragma unroll
  for (int r = 0; r < 8; ++r) {
    const int s = rbase + r;
    float* orow = attn + (size_t)(b * Ssz + s) * Dsz + h * HDsz;
    orow[d0 + n] = acc0[r];
    orow[d1 + n] = acc1[r];
  }
}

extern "C" void kernel_launch(void* const* d_in, const int* in_sizes, int n_in,
                              void* d_out, int out_size, void* d_ws, size_t ws_size,
                              hipStream_t stream) {
  const float* x       = (const float*)d_in[0];
  const float* cosb    = (const float*)d_in[1];
  const float* sinb    = (const float*)d_in[2];
  const float* mask    = (const float*)d_in[3];
  const int*   idx     = (const int*)d_in[4];
  const float* cache_k = (const float*)d_in[5];
  const float* cache_v = (const float*)d_in[6];
  const float* wq      = (const float*)d_in[7];
  const float* wk      = (const float*)d_in[8];
  const float* wv      = (const float*)d_in[9];
  const float* wo      = (const float*)d_in[10];

  float* out_proj = (float*)d_out;
  float* kout = out_proj + OUT_ELEMS;
  float* vout = kout + CACHE_ELEMS;

  // workspace carve-up
  float* xq     = (float*)d_ws;
  float* xk     = xq + OUT_ELEMS;
  float* xv     = xk + OUT_ELEMS;
  float* scores = xv + OUT_ELEMS;                        // B*H*S*T f32
  const size_t SCORE_ELEMS = (size_t)Bsz * Hn * Ssz * Tsz;
  _Float16* probs = (_Float16*)(scores + SCORE_ELEMS);   // B*H*S*T f16
  float* attn   = (float*)(probs + SCORE_ELEMS);         // [128,4096] f32

  // 1) QKV projections (WMMA GEMMs, async-LDS weight staging)
  gemm_xwT_wmma<<<dim3(Dsz / 16), dim3(256), 0, stream>>>(x, wq, xq, Dsz, Dsz);
  gemm_xwT_wmma<<<dim3(Dsz / 16), dim3(256), 0, stream>>>(x, wk, xk, Dsz, Dsz);
  gemm_xwT_wmma<<<dim3(Dsz / 16), dim3(256), 0, stream>>>(x, wv, xv, Dsz, Dsz);

  // 2) RoPE on xq, xk
  rope_kernel<<<dim3(2048), dim3(256), 0, stream>>>(xq, xk, cosb, sinb);

  // 3) stream caches to output (non-temporal) + scatter fresh K/V rows
  copy_caches<<<dim3(8192), dim3(256), 0, stream>>>(
      (const v4f*)cache_k, (const v4f*)cache_v,
      (v4f*)kout, (v4f*)vout, (long)(CACHE_ELEMS / 4));
  scatter_kv<<<dim3(512), dim3(256), 0, stream>>>(
      (const v4f*)xk, (const v4f*)xv, idx, (v4f*)kout, (v4f*)vout);

  // 4) attention scores (reads updated K cache from d_out)
  attn_scores_wmma<<<dim3(Tsz / 64, Bsz * Hn), dim3(128), 0, stream>>>(xq, kout, mask, scores);

  // 5) softmax
  softmax_rows<<<dim3(Bsz * Hn * Ssz), dim3(256), 0, stream>>>(scores, probs);

  // 6) probs @ V
  attn_pv_wmma<<<dim3(Bsz * Hn), dim3(128), 0, stream>>>(probs, vout, attn);

  // 7) output projection
  gemm_xwT_wmma<<<dim3(Dsz / 16), dim3(256), 0, stream>>>(attn, wo, out_proj, Dsz, Dsz);
}
